// RegionAttention_67869073211924
// MI455X (gfx1250) — matrix operand order repaired and checked by reading, not compiled, not run on hardware
//
#include <hip/hip_runtime.h>
#include <hip/hip_bf16.h>

// ---------------------------------------------------------------------------
// RegionAttention on gfx1250 (MI455X): flash-attention with v_wmma_f32_16x16x32_f16
//  - mask (134 MB) streamed once, scores never materialized
//  - head_dim 24 padded to 32 -> one WMMA covers full K of QK^T
//  - K/V tiles staged into LDS with global_load_async_to_lds_b128 (ASYNCcnt),
//    double-buffered, shared by all 4 waves of the workgroup (4x less L2 traffic)
// ---------------------------------------------------------------------------

typedef __attribute__((ext_vector_type(16))) _Float16 v16h;
typedef __attribute__((ext_vector_type(8)))  _Float16 v8h;
typedef __attribute__((ext_vector_type(8)))  float    v8f;

#define WMMA16x16x32(a, b, c) \
  __builtin_amdgcn_wmma_f32_16x16x32_f16(false, (a), false, (b), (short)0, (c), false, false)

namespace ra {
constexpr int   B   = 2;
constexpr int   C   = 192;
constexpr int   L   = 64 * 64;          // 4096
constexpr int   NH  = 8;
constexpr int   HD  = 24;               // head dim
constexpr int   HDP = 32;               // padded head dim
constexpr int   C3  = 3 * C;            // 576
constexpr float SCALE = 0.35355339059327373f;  // num_heads^-0.5 (per reference)
}

// --- A operand (16x32 f16, MxK): lane holds 16 contiguous K values starting at 2*c0, c0=8*(lane>=16)
__device__ __forceinline__ v16h make_a(const _Float16* row, int c0) {
  v8h lo = *(const v8h*)(row + 2 * c0);
  v8h hi = *(const v8h*)(row + 2 * c0 + 8);
  v16h a;
#pragma unroll
  for (int i = 0; i < 8; ++i) { a[i] = lo[i]; a[i + 8] = hi[i]; }
  return a;
}
// --- B operand (32x16 f16, KxN): lane holds 16 contiguous K values starting at koff=16*(lane>=16)
__device__ __forceinline__ v16h make_b(const _Float16* p) {
  v8h lo = *(const v8h*)(p);
  v8h hi = *(const v8h*)(p + 8);
  v16h b;
#pragma unroll
  for (int i = 0; i < 8; ++i) { b[i] = lo[i]; b[i + 8] = hi[i]; }
  return b;
}

// async 16B/lane copy global -> LDS, tracked by ASYNCcnt (gfx1250)
__device__ __forceinline__ void async_copy_b128(unsigned int lds_byte, const void* gaddr) {
  asm volatile("global_load_async_to_lds_b128 %0, %1, off"
               : : "v"(lds_byte), "v"(gaddr) : "memory");
}
__device__ __forceinline__ unsigned int lds_addr32(const void* p) {
  return (unsigned int)(unsigned long long)p;   // low 32 bits = LDS byte offset
}

// ---------------------------------------------------------------------------
// prep kernels
// ---------------------------------------------------------------------------
__global__ void k_zero_u32(unsigned int* p, int n) {
  for (int i = blockIdx.x * blockDim.x + threadIdx.x; i < n; i += gridDim.x * blockDim.x)
    p[i] = 0u;
}

__global__ void k_cvt_weights(const float* __restrict__ w_qkv, const float* __restrict__ w_proj,
                              _Float16* __restrict__ wqkv_h, _Float16* __restrict__ wproj_h) {
  const int n_qkv  = ra::C3 * ra::C;
  const int n_total = n_qkv + ra::C * ra::C;
  for (int i = blockIdx.x * blockDim.x + threadIdx.x; i < n_total; i += gridDim.x * blockDim.x) {
    if (i < n_qkv) wqkv_h[i] = (_Float16)w_qkv[i];
    else           wproj_h[i - n_qkv] = (_Float16)w_proj[i - n_qkv];
  }
}

// x [B,C,H,W] f32 (channel-major) -> xT [B*L, C] f16 (pixel-major rows)
__global__ void k_transpose_x(const float* __restrict__ x, _Float16* __restrict__ xT) {
  const int n = ra::B * ra::C * ra::L;
  for (int i = blockIdx.x * blockDim.x + threadIdx.x; i < n; i += gridDim.x * blockDim.x) {
    int b = i / (ra::C * ra::L);
    int r = i - b * (ra::C * ra::L);
    int c = r / ra::L;
    int p = r - c * ra::L;
    xT[((size_t)(b * ra::L + p)) * ra::C + c] = (_Float16)x[i];
  }
}

// ---------------------------------------------------------------------------
// Phase 1: QKV projection GEMM  [B*L,192] x [192,576]^T  (one 16x16 tile / wave)
// ---------------------------------------------------------------------------
__global__ __launch_bounds__(128) void k_gemm_qkv(
    const _Float16* __restrict__ xT, const _Float16* __restrict__ wqkv_h,
    const float* __restrict__ b_qkv,
    _Float16* __restrict__ Qp, _Float16* __restrict__ Kp, _Float16* __restrict__ Vt) {
  using namespace ra;
  const int wave = threadIdx.x >> 5, lane = threadIdx.x & 31;
  const int tile = blockIdx.x * 4 + wave;
  const int Mt = (B * L) / 16;                 // 512
  const int mt = tile % Mt, nt = tile / Mt;    // nt in [0,36)
  const int p0 = mt * 16, o0 = nt * 16;
  const int nl = lane & 15, hi = lane >> 4;
  const int c0 = hi * 8, koff = hi * 16;

  const _Float16* arow = xT     + (size_t)(p0 + nl) * C;
  const _Float16* brow = wqkv_h + (size_t)(o0 + nl) * C;
  v8f acc = {};
#pragma unroll
  for (int kb = 0; kb < C; kb += 32) {
    v16h a = make_a(arow + kb, c0);
    v16h b = make_b(brow + kb + koff);
    acc = WMMA16x16x32(a, b, acc);
  }

  const int o = o0 + nl;
  const float bias = b_qkv[o];
  const int sec = o0 / C;                      // wave-uniform: 0=Q,1=K,2=V
  const int cc = o - sec * C;
  const int head = cc / HD, dim = cc - head * HD;
  const int bb = p0 / L;                       // tile never straddles batch
  const int pl = p0 - bb * L;

  if (sec == 0) {
    _Float16* dst = Qp + (size_t)(bb * NH + head) * L * HDP;
#pragma unroll
    for (int v = 0; v < 8; ++v) {
      int p = pl + v + hi * 8;
      dst[(size_t)p * HDP + dim] = (_Float16)((acc[v] + bias) * SCALE);
    }
  } else if (sec == 1) {
    _Float16* dst = Kp + (size_t)(bb * NH + head) * L * HDP;
#pragma unroll
    for (int v = 0; v < 8; ++v) {
      int p = pl + v + hi * 8;
      dst[(size_t)p * HDP + dim] = (_Float16)(acc[v] + bias);
    }
  } else {
    _Float16* dst = Vt + ((size_t)(bb * NH + head) * HDP + dim) * L;
    v8h pk;
#pragma unroll
    for (int v = 0; v < 8; ++v) pk[v] = (_Float16)(acc[v] + bias);
    *(v8h*)(dst + pl + hi * 8) = pk;           // 8 consecutive pixels
  }
}

// ---------------------------------------------------------------------------
// Phase 2: flash attention. 1 wave = 16 query rows; 4 waves/WG share LDS-staged K/V.
// ---------------------------------------------------------------------------
__global__ __launch_bounds__(128) void k_attn(
    const _Float16* __restrict__ Qp, const _Float16* __restrict__ Kp,
    const _Float16* __restrict__ Vt, const float* __restrict__ mask,
    _Float16* __restrict__ Oh) {
  using namespace ra;
  __shared__ __align__(128) _Float16 Kbuf[2][32 * 32];  // [key][dim32], 2 KB per buffer
  __shared__ __align__(128) _Float16 Vbuf[2][32 * 32];  // [dim32][key], 2 KB per buffer
  __shared__ __align__(64)  _Float16 Ps[4][16 * 32];    // per-wave P staging (D->A relayout)

  const int wave = threadIdx.x >> 5, lane = threadIdx.x & 31;
  const int head = blockIdx.y, b = blockIdx.z;
  const int q0 = blockIdx.x * 64 + wave * 16;
  const size_t bh = (size_t)(b * NH + head);

  const _Float16* Qb = Qp + bh * L * HDP;
  const _Float16* Kb = Kp + bh * L * HDP;
  const _Float16* Vb = Vt + bh * HDP * L;
  const float*    Mb = mask + (size_t)b * L * L;

  const int nl = lane & 15, hi = lane >> 4;
  const int c0 = hi * 8, koff = hi * 16;

  const v16h aq = make_a(Qb + (size_t)(q0 + nl) * HDP, c0);   // Q rows, resident whole loop

  float mi[8], li[8];
  v8f acc0 = {}, acc1 = {};
#pragma unroll
  for (int v = 0; v < 8; ++v) { mi[v] = -3.0e38f; li[v] = 0.f; }

  _Float16* Pw = &Ps[wave][0];
  const v8f zc = {};

  // stage one 32-key block (K: 2KB, V: 2KB) -- 8 async b128 ops split over 4 waves
  auto stage = [&](int buf, int kb) {
    if (wave < 2) {          // K tile: contiguous 2 KB, linear copy
#pragma unroll
      for (int j = 0; j < 2; ++j) {
        int chunk = wave * 2 + j;                       // 0..3, 512 B each
        const char* g = (const char*)(Kb + (size_t)kb * HDP) + chunk * 512 + lane * 16;
        unsigned int l = lds_addr32(&Kbuf[buf][0]) + chunk * 512 + lane * 16;
        async_copy_b128(l, g);
      }
    } else {                 // V tile: 32 rows of 64 B, row stride L halves in global
#pragma unroll
      for (int j = 0; j < 2; ++j) {
        int chunk = (wave - 2) * 2 + j;                 // 0..3 -> dims 8*chunk..8*chunk+7
        int dim = chunk * 8 + (lane >> 2);
        int ch  = lane & 3;
        const char* g = (const char*)(Vb + (size_t)dim * L + kb) + ch * 16;
        unsigned int l = lds_addr32(&Vbuf[buf][0]) + dim * 64 + ch * 16;
        async_copy_b128(l, g);
      }
    }
  };

  stage(0, 0);   // prologue: 2 async ops outstanding per wave

  for (int kb = 0, it = 0; kb < L; kb += 32, ++it) {
    const int cur = it & 1, nxt = cur ^ 1;
    const int kb2 = (kb + 32 < L) ? kb + 32 : 0;        // wrap: uniform 2-op stage every iter

    __syncthreads();                                    // buffer 'nxt' free to overwrite
    stage(nxt, kb2);
    asm volatile("s_wait_asynccnt 0x2" ::: "memory");   // the 2 older ops (buf 'cur') done
    __syncthreads();                                    // 'cur' visible to all waves

    // S = Q K^T (pre-scaled Q), two 16-key tiles, B-operands from LDS
    v16h bk0 = make_b(&Kbuf[cur][(size_t)nl * HDP + koff]);
    v16h bk1 = make_b(&Kbuf[cur][(size_t)(16 + nl) * HDP + koff]);
    v8f s0 = WMMA16x16x32(aq, bk0, zc);
    v8f s1 = WMMA16x16x32(aq, bk1, zc);

    const float* mrow = Mb + (size_t)(q0 + hi * 8) * L + kb + nl;
    __builtin_prefetch(mrow + 32, 0, 1);                // global_prefetch next mask block
#pragma unroll
    for (int v = 0; v < 8; ++v) {
      const float* mr = mrow + (size_t)v * L;
      float t0 = s0[v] + mr[0];
      float t1 = s1[v] + mr[16];
      // row max over 32 keys (cols live across each 16-lane half)
      float bm = fmaxf(t0, t1);
      bm = fmaxf(bm, __shfl_xor(bm, 1));
      bm = fmaxf(bm, __shfl_xor(bm, 2));
      bm = fmaxf(bm, __shfl_xor(bm, 4));
      bm = fmaxf(bm, __shfl_xor(bm, 8));
      float nm   = fmaxf(mi[v], bm);
      float corr = __expf(mi[v] - nm);
      mi[v] = nm;
      float e0 = __expf(t0 - nm);
      float e1 = __expf(t1 - nm);
      float rs = e0 + e1;
      rs += __shfl_xor(rs, 1);
      rs += __shfl_xor(rs, 2);
      rs += __shfl_xor(rs, 4);
      rs += __shfl_xor(rs, 8);
      li[v]   = li[v] * corr + rs;
      acc0[v] *= corr;
      acc1[v] *= corr;
      // stage P in plain [m][k] so it can be re-read in A-operand layout
      int m = v + hi * 8;
      Pw[m * 32 + nl]      = (_Float16)e0;
      Pw[m * 32 + 16 + nl] = (_Float16)e1;
    }

    // PV: A = P (16x32), B = V^T rows from LDS, two dim tiles (0-15, 16-31)
    v16h ap  = make_a(Pw + nl * 32, c0);
    v16h bv0 = make_b(&Vbuf[cur][(size_t)nl * HDP + koff]);
    v16h bv1 = make_b(&Vbuf[cur][(size_t)(16 + nl) * HDP + koff]);
    acc0 = WMMA16x16x32(ap, bv0, acc0);
    acc1 = WMMA16x16x32(ap, bv1, acc1);
  }

  // drain the wrap-around prefetch before the wave exits
  asm volatile("s_wait_asynccnt 0x0" ::: "memory");

  // normalize and write O as [B*L, 192] f16 (channel = head*24 + dim)
  const int chbase = head * HD;
#pragma unroll
  for (int v = 0; v < 8; ++v) {
    int q = q0 + v + hi * 8;
    _Float16* orow = Oh + ((size_t)b * L + q) * C + chbase;
    float inv = 1.0f / li[v];
    orow[nl] = (_Float16)(acc0[v] * inv);
    if (nl < 8) orow[16 + nl] = (_Float16)(acc1[v] * inv);   // dims 16..23 only
  }
}

// ---------------------------------------------------------------------------
// Phase 3: output projection GEMM [B*L,192] x [192,192]^T -> d_out [B,C,H,W] f32
// ---------------------------------------------------------------------------
__global__ __launch_bounds__(128) void k_gemm_proj(
    const _Float16* __restrict__ Oh, const _Float16* __restrict__ wproj_h,
    const float* __restrict__ b_proj, float* __restrict__ out) {
  using namespace ra;
  const int wave = threadIdx.x >> 5, lane = threadIdx.x & 31;
  const int tile = blockIdx.x * 4 + wave;
  const int Mt = (B * L) / 16;                 // 512
  const int mt = tile % Mt, nt = tile / Mt;    // nt in [0,12)
  const int p0 = mt * 16, o0 = nt * 16;
  const int nl = lane & 15, hi = lane >> 4;
  const int c0 = hi * 8, koff = hi * 16;

  const _Float16* arow = Oh      + (size_t)(p0 + nl) * C;
  const _Float16* brow = wproj_h + (size_t)(o0 + nl) * C;
  v8f acc = {};
#pragma unroll
  for (int kb = 0; kb < C; kb += 32) {
    v16h a = make_a(arow + kb, c0);
    v16h b = make_b(brow + kb + koff);
    acc = WMMA16x16x32(a, b, acc);
  }

  const int o = o0 + nl;
  const float bias = b_proj[o];
  const int bb = p0 / L;
  const int pl = p0 - bb * L;
  float* dst = out + ((size_t)bb * C + o) * L + pl + hi * 8;  // 8 consecutive pixels per lane
  *(float4*)(dst)     = make_float4(acc[0] + bias, acc[1] + bias, acc[2] + bias, acc[3] + bias);
  *(float4*)(dst + 4) = make_float4(acc[4] + bias, acc[5] + bias, acc[6] + bias, acc[7] + bias);
}

// ---------------------------------------------------------------------------
// launcher
// ---------------------------------------------------------------------------
extern "C" void kernel_launch(void* const* d_in, const int* in_sizes, int n_in,
                              void* d_out, int out_size, void* d_ws, size_t ws_size,
                              hipStream_t stream) {
  using namespace ra;
  const float* x      = (const float*)d_in[0];
  const float* mask   = (const float*)d_in[1];
  const float* w_qkv  = (const float*)d_in[2];
  const float* b_qkv  = (const float*)d_in[3];
  const float* w_proj = (const float*)d_in[4];
  const float* b_proj = (const float*)d_in[5];
  float* out = (float*)d_out;

  // workspace carve-up (f16 buffers), ~18.3 MB total
  char* w = (char*)d_ws;
  _Float16* xT      = (_Float16*)w;  w += (size_t)B * L * C * 2;            // 3 MB
  _Float16* wqkv_h  = (_Float16*)w;  w += (size_t)C3 * C * 2;               // 216 KB
  _Float16* wproj_h = (_Float16*)w;  w += (size_t)C * C * 2;                // 72 KB
  _Float16* Qp      = (_Float16*)w;  w += (size_t)B * NH * L * HDP * 2;     // 4 MB
  _Float16* Kp      = (_Float16*)w;  w += (size_t)B * NH * L * HDP * 2;     // 4 MB
  _Float16* Vt      = (_Float16*)w;  w += (size_t)B * NH * L * HDP * 2;     // 4 MB
  _Float16* Oh      = (_Float16*)w;  w += (size_t)B * L * C * 2;            // 3 MB

  // zero padded Q/K/Vt (dims 24..31 must be 0 for the X32 WMMA K-padding)
  const int zero_words = (3 * B * NH * L * HDP * 2) / 4;
  k_zero_u32<<<4096, 256, 0, stream>>>((unsigned int*)Qp, zero_words);
  k_cvt_weights<<<576, 256, 0, stream>>>(w_qkv, w_proj, wqkv_h, wproj_h);
  k_transpose_x<<<6144, 256, 0, stream>>>(x, xT);

  // QKV projection: 512 x 36 tiles, 4 waves/block
  k_gemm_qkv<<<(512 * 36) / 4, 128, 0, stream>>>(xT, wqkv_h, b_qkv, Qp, Kp, Vt);

  // flash attention: grid (qblocks=64, heads=8, batch=2), 128 threads
  k_attn<<<dim3(64, NH, B), 128, 0, stream>>>(Qp, Kp, Vt, mask, Oh);

  // output projection: 512 x 12 tiles
  k_gemm_proj<<<(512 * 12) / 4, 128, 0, stream>>>(Oh, wproj_h, b_proj, out);

  (void)in_sizes; (void)n_in; (void)out_size; (void)ws_size;
}
